// BiaffineCRFClassifier_60421599920243
// MI455X (gfx1250) — matrix operand order repaired and checked by reading, not compiled, not run on hardware
//
#include <hip/hip_runtime.h>
#include <hip/hip_bf16.h>
#include <stdint.h>

#define BB 8
#define NN 256
#define DD 1024
#define HH 300
#define LL 96
#define KP 320            // padded feature dim (H+1=301 -> 320, multiple of 32)
#define MROWS (BB*NN)     // 2048

typedef __attribute__((ext_vector_type(16))) _Float16 v16h;
typedef __attribute__((ext_vector_type(8)))  _Float16 v8h;
typedef __attribute__((ext_vector_type(8)))  float    v8f;
typedef __attribute__((ext_vector_type(4)))  unsigned int v4u;
typedef __attribute__((ext_vector_type(8)))  int      v8i;
typedef __attribute__((ext_vector_type(4)))  int      v4i;

// ---- WMMA operand loaders (CDNA5 16x16x32 f16 layouts, wave32) ----
// A (16x32, MxK): lane m=lane&15 holds row m; lanes<16 take K {0..7,16..23},
// lanes>=16 take K {8..15,24..31}.
__device__ __forceinline__ v16h loadA16(const _Float16* p, int lane) {
  const int sel = ((lane >> 4) & 1) << 3;          // 0 or 8 halfs
  v8h lo = *(const v8h*)(p + sel);
  v8h hi = *(const v8h*)(p + 16 + sel);
  v16h r;
  #pragma unroll
  for (int i = 0; i < 8; ++i) { r[i] = lo[i]; r[i + 8] = hi[i]; }
  return r;
}
// B (32x16, KxN): lane n=lane&15 holds column n; lanes<16 hold K {0..7,16..23},
// lanes>=16 hold K {8..15,24..31} (caller adds the +8 half offset).
__device__ __forceinline__ v16h loadB16(const _Float16* p) {
  v8h lo = *(const v8h*)(p);
  v8h hi = *(const v8h*)(p + 16);
  v16h r;
  #pragma unroll
  for (int i = 0; i < 8; ++i) { r[i] = lo[i]; r[i + 8] = hi[i]; }
  return r;
}

// ---------------- pack / convert kernels ----------------
__global__ void k_cvt_f16(const float* __restrict__ s, _Float16* __restrict__ d, int n) {
  for (int i = blockIdx.x * blockDim.x + threadIdx.x; i < n; i += gridDim.x * blockDim.x)
    d[i] = (_Float16)s[i];
}

// WT[n*DD + k] = (n < H) ? W[k*H + n] : 0   (n < KP)
__global__ void k_pack_WT(const float* __restrict__ Wh, const float* __restrict__ Wd,
                          _Float16* __restrict__ WhT, _Float16* __restrict__ WdT) {
  const float* W = blockIdx.y ? Wd : Wh;
  _Float16* WT = blockIdx.y ? WdT : WhT;
  const int total = KP * DD;
  for (int i = blockIdx.x * blockDim.x + threadIdx.x; i < total; i += gridDim.x * blockDim.x) {
    int n = i / DD, k = i % DD;
    WT[i] = (n < HH) ? (_Float16)W[k * HH + n] : (_Float16)0.f;
  }
}

// WpT[(l*KP + j)*KP + i] = (i<301 && j<301) ? Wbi[(l*301 + i)*301 + j] : 0
__global__ void k_pack_WbiT(const float* __restrict__ Wbi, _Float16* __restrict__ WpT) {
  const size_t total = (size_t)LL * KP * KP;
  for (size_t idx = blockIdx.x * (size_t)blockDim.x + threadIdx.x; idx < total;
       idx += (size_t)gridDim.x * blockDim.x) {
    int i = (int)(idx % KP);
    int j = (int)((idx / KP) % KP);
    int l = (int)(idx / ((size_t)KP * KP));
    _Float16 v = (_Float16)0.f;
    if (i < HH + 1 && j < HH + 1) v = (_Float16)Wbi[((size_t)l * 301 + i) * 301 + j];
    WpT[idx] = v;
  }
}

// ---------------- projection: x/y = relu(input@W + b) ++ [1], f16, padded ----
__global__ void __launch_bounds__(32)
k_proj_wmma(const _Float16* __restrict__ Ain, const _Float16* __restrict__ WhT,
            const _Float16* __restrict__ WdT, const float* __restrict__ bh,
            const float* __restrict__ bd, _Float16* __restrict__ xout,
            _Float16* __restrict__ yout) {
  const int mt = blockIdx.x;                // 0..127
  const int nt = blockIdx.y;                // 0..19
  const int which = blockIdx.z;             // 0=h, 1=d
  const _Float16* WT = which ? WdT : WhT;
  const float* bias  = which ? bd  : bh;
  _Float16* outp     = which ? yout : xout;
  const int lane = threadIdx.x;
  const int n = lane & 15;
  const int kh = (lane >= 16) ? 8 : 0;
  const _Float16* arow = Ain + ((size_t)mt * 16 + n) * DD;
  const _Float16* brow = WT + ((size_t)nt * 16 + n) * DD;
  v8f acc = {};
  for (int k = 0; k < DD; k += 32) {
    v16h a  = loadA16(arow + k, lane);
    v16h bm = loadB16(brow + k + kh);
    acc = __builtin_amdgcn_wmma_f32_16x16x32_f16(false, a, false, bm, (short)0, acc, false, false);
  }
  const int col = nt * 16 + n;
  const int rbase = (lane < 16) ? 0 : 8;
  #pragma unroll
  for (int v = 0; v < 8; ++v) {
    const int row = mt * 16 + rbase + v;
    float val;
    if (col < HH)            val = fmaxf(acc[v] + bias[col], 0.f);
    else if (col == HH)      val = 1.0f;
    else                     val = 0.f;
    outp[(size_t)row * KP + col] = (_Float16)val;
  }
}

// ---------------- fused biaffine: S[b,x,y,l] = (X W_l Y^T) ----------------
__global__ void __launch_bounds__(256)
k_biaffine(const _Float16* __restrict__ xh, const _Float16* __restrict__ yh,
           const _Float16* __restrict__ WpT, float* __restrict__ emis) {
  __shared__ __align__(16) _Float16 Xs[16 * KP];   // 10240 B
  __shared__ __align__(16) _Float16 Us[16 * KP];   // 10240 B
  const int xt = blockIdx.x;   // 0..15 (16-row strip)
  const int l  = blockIdx.y;   // 0..95
  const int b  = blockIdx.z;   // 0..7
  const int tid = threadIdx.x;
  const int lane = tid & 31;
  const int w = tid >> 5;

  // ---- TDM: DMA the contiguous 16x320-half X strip into LDS (wave 0 only) ----
  if (w == 0) {
    const unsigned long long gaddr =
        (unsigned long long)(const void*)(xh + ((size_t)(b * NN) + xt * 16) * KP);
    const unsigned int laddr = (unsigned int)(size_t)(void*)Xs;
    // D# group0: count=1 | lds_addr | global_addr[56:0] | type=2
    v4u g0;
    g0[0] = 1u;
    g0[1] = laddr;
    g0[2] = (unsigned int)gaddr;
    g0[3] = (unsigned int)((gaddr >> 32) & 0x01FFFFFFu) | (2u << 30);
    // D# group1: 1-D copy, data_size=8B (code 3), dim0=tile0=stride0=1280 (=10240 B)
    const unsigned int elems = (16u * KP * 2u) / 8u;   // 1280
    v8i g1;
    g1[0] = (int)(3u << 16);                       // workgroup_mask=0, data_size=3
    g1[1] = (int)((elems & 0xFFFFu) << 16);        // tensor_dim0[15:0] @ bits 63:48
    g1[2] = (int)((elems >> 16) | (1u << 16));     // tensor_dim0[31:16], tensor_dim1=1
    g1[3] = (int)((elems & 0xFFFFu) << 16);        // tile_dim0 @ bits 127:112
    g1[4] = 1;                                     // tile_dim1=1, tile_dim2=0
    g1[5] = (int)elems;                            // tensor_dim0_stride[31:0]
    g1[6] = 0;                                     // stride0[47:32], stride1[15:0]
    g1[7] = 0;
    v4i gz4 = {};
    v8i gz8 = {};
    __builtin_amdgcn_tensor_load_to_lds(g0, g1, gz4, gz4, gz8, 0);
    __builtin_amdgcn_s_wait_tensorcnt(0);
  }
  __syncthreads();

  const int n = lane & 15;
  const int kh = (lane >= 16) ? 8 : 0;
  const int rbase = (lane < 16) ? 0 : 8;

  // phase 1: Us(16x320) = Xs(16x320) @ W_l(320x320); B from WpT (j-major, i-contig)
  for (int nt = w; nt < KP / 16; nt += 8) {
    v8f acc = {};
    const _Float16* arow = Xs + n * KP;
    const _Float16* brow = WpT + ((size_t)l * KP + nt * 16 + n) * KP;
    if (nt + 8 < KP / 16)
      __builtin_prefetch(brow + 16 * KP, 0, 0);    // warm next tile's B rows in L2
    #pragma unroll
    for (int k = 0; k < KP; k += 32) {
      v16h a  = loadA16(arow + k, lane);
      v16h bm = loadB16(brow + k + kh);
      acc = __builtin_amdgcn_wmma_f32_16x16x32_f16(false, a, false, bm, (short)0, acc, false, false);
    }
    const int col = nt * 16 + n;
    #pragma unroll
    for (int v = 0; v < 8; ++v) Us[(rbase + v) * KP + col] = (_Float16)acc[v];
  }
  __syncthreads();

  // phase 2: S(16x256) = Us(16x320) @ Y^T; B lane = row of Y (j-contiguous)
  for (int yt = w; yt < NN / 16; yt += 8) {
    v8f acc = {};
    const _Float16* arow = Us + n * KP;
    const _Float16* brow = yh + ((size_t)(b * NN) + yt * 16 + n) * KP;
    if (yt + 8 < NN / 16)
      __builtin_prefetch(brow + 8 * 16 * KP, 0, 0);
    #pragma unroll
    for (int k = 0; k < KP; k += 32) {
      v16h a  = loadA16(arow + k, lane);
      v16h bm = loadB16(brow + k + kh);
      acc = __builtin_amdgcn_wmma_f32_16x16x32_f16(false, a, false, bm, (short)0, acc, false, false);
    }
    const int y = yt * 16 + n;
    #pragma unroll
    for (int v = 0; v < 8; ++v) {
      const int x = xt * 16 + rbase + v;
      emis[(((size_t)(b * NN) + x) * NN + y) * LL + l] = acc[v];
    }
  }
}

// ---------------- in-place log_softmax over L=96 (one wave / row) ----------
__global__ void __launch_bounds__(256) k_logsoftmax(float* __restrict__ emis) {
  const int row = blockIdx.x * 8 + (threadIdx.x >> 5);
  const int lane = threadIdx.x & 31;
  float* p = emis + (size_t)row * LL;
  float v0 = p[lane], v1 = p[lane + 32], v2 = p[lane + 64];
  float m = fmaxf(v0, fmaxf(v1, v2));
  #pragma unroll
  for (int off = 16; off > 0; off >>= 1) m = fmaxf(m, __shfl_xor(m, off));
  float s = __expf(v0 - m) + __expf(v1 - m) + __expf(v2 - m);
  #pragma unroll
  for (int off = 16; off > 0; off >>= 1) s += __shfl_xor(s, off);
  const float lse = m + __logf(s);
  p[lane] = v0 - lse; p[lane + 32] = v1 - lse; p[lane + 64] = v2 - lse;
}

// ---------------- CRF numerator (one wave / row) ----------------
__global__ void __launch_bounds__(256)
k_numerator(const float* __restrict__ emis, const unsigned char* __restrict__ wam,
            const int* __restrict__ labels, const float* __restrict__ trans,
            const float* __restrict__ startv, const float* __restrict__ endv,
            float* __restrict__ num) {
  const int r = blockIdx.x * 8 + (threadIdx.x >> 5);
  const int lane = threadIdx.x & 31;
  const int b = r >> 8;
  const bool rowvalid = wam[b * NN + (r & 255)] != 0;
  const int* tags = labels + (size_t)r * NN;
  const float* erow = emis + (size_t)r * NN * LL;
  float acc = 0.f;
  int cnt = 0;
  for (int t = lane; t < NN; t += 32) {
    if (t >= 1 && rowvalid && (wam[b * NN + t] != 0)) {
      const int tp = tags[t - 1], tc = tags[t];
      acc += trans[tp * LL + tc] + erow[(size_t)t * LL + tc];
      cnt += 1;
    }
  }
  #pragma unroll
  for (int off = 16; off > 0; off >>= 1) { acc += __shfl_xor(acc, off); cnt += __shfl_xor(cnt, off); }
  if (lane == 0) {
    const int t0 = tags[0];
    float v = startv[t0] + erow[t0] + acc;
    const int last_idx = cnt;                 // (1 + cnt) - 1, mask is a prefix
    v += endv[tags[last_idx]];
    num[r] = v;
  }
}

// ---------------- CRF forward scan (one block / row, trans in LDS) --------
__global__ void __launch_bounds__(128)
k_crf_forward(const float* __restrict__ emis, const unsigned char* __restrict__ wam,
              const float* __restrict__ trans, const float* __restrict__ startv,
              const float* __restrict__ endv, const float* __restrict__ num,
              float* __restrict__ ll) {
  __shared__ float trS[LL * LL];   // 36 KB
  __shared__ float alphaS[LL];
  __shared__ float red[128];
  const int r = blockIdx.x;
  const int tid = threadIdx.x;
  const int b = r >> 8;
  const bool rowvalid = wam[b * NN + (r & 255)] != 0;
  for (int i = tid; i < LL * LL; i += 128) trS[i] = trans[i];
  const float* erow = emis + (size_t)r * NN * LL;
  if (tid < LL) alphaS[tid] = startv[tid] + erow[tid];
  __syncthreads();
  const int j = tid;
  for (int t = 1; t < NN; ++t) {
    const bool mt = rowvalid && (wam[b * NN + t] != 0);
    if (mt) {                                   // uniform branch per block
      float nxt = 0.f;
      if (j < LL) {
        float m = -3.4e38f;
        for (int i = 0; i < LL; ++i) m = fmaxf(m, alphaS[i] + trS[i * LL + j]);
        float s = 0.f;
        for (int i = 0; i < LL; ++i) s += __expf(alphaS[i] + trS[i * LL + j] - m);
        nxt = erow[(size_t)t * LL + j] + m + __logf(s);
      }
      __syncthreads();
      if (j < LL) alphaS[j] = nxt;
      __syncthreads();
    }
  }
  const float v = (j < LL) ? alphaS[j] + endv[j] : -3.4e38f;
  red[tid] = v;
  __syncthreads();
  for (int off = 64; off > 0; off >>= 1) { if (tid < off) red[tid] = fmaxf(red[tid], red[tid + off]); __syncthreads(); }
  const float m = red[0];
  __syncthreads();
  red[tid] = (j < LL) ? __expf(v - m) : 0.f;
  __syncthreads();
  for (int off = 64; off > 0; off >>= 1) { if (tid < off) red[tid] += red[tid + off]; __syncthreads(); }
  if (tid == 0) {
    const float logZ = m + __logf(red[0]);
    ll[r] = rowvalid ? (num[r] - logZ) : 0.f;
  }
}

// ---------------- final reduction ----------------
__global__ void __launch_bounds__(256) k_finalize(const float* __restrict__ ll, float* __restrict__ out) {
  __shared__ float red[256];
  float s = 0.f;
  for (int i = threadIdx.x; i < MROWS; i += 256) s += ll[i];
  red[threadIdx.x] = s;
  __syncthreads();
  for (int off = 128; off > 0; off >>= 1) { if (threadIdx.x < off) red[threadIdx.x] += red[threadIdx.x + off]; __syncthreads(); }
  if (threadIdx.x == 0) out[0] = -red[0];
}

extern "C" void kernel_launch(void* const* d_in, const int* in_sizes, int n_in,
                              void* d_out, int out_size, void* d_ws, size_t ws_size,
                              hipStream_t stream) {
  (void)in_sizes; (void)n_in; (void)out_size; (void)ws_size;
  const float* input          = (const float*)d_in[0];
  const unsigned char* wam    = (const unsigned char*)d_in[1];   // jnp.bool_
  const int* labels           = (const int*)d_in[2];
  const float* W_h            = (const float*)d_in[3];
  const float* b_h            = (const float*)d_in[4];
  const float* W_d            = (const float*)d_in[5];
  const float* b_d            = (const float*)d_in[6];
  const float* W_bi           = (const float*)d_in[7];
  const float* startv         = (const float*)d_in[8];
  const float* endv           = (const float*)d_in[9];
  const float* transv         = (const float*)d_in[10];
  float* out = (float*)d_out;

  char* ws = (char*)d_ws;
  size_t off = 0;
  auto carve = [&](size_t bytes) { char* p = ws + off; off += (bytes + 255) & ~(size_t)255; return p; };
  _Float16* inph = (_Float16*)carve((size_t)MROWS * DD * 2);       // 4 MB
  _Float16* WhT  = (_Float16*)carve((size_t)KP * DD * 2);          // 0.64 MB
  _Float16* WdT  = (_Float16*)carve((size_t)KP * DD * 2);
  _Float16* xh   = (_Float16*)carve((size_t)MROWS * KP * 2);       // 1.3 MB
  _Float16* yh   = (_Float16*)carve((size_t)MROWS * KP * 2);
  _Float16* WpT  = (_Float16*)carve((size_t)LL * KP * KP * 2);     // 19.7 MB
  float* emis    = (float*)carve((size_t)MROWS * NN * LL * 4);     // 201 MB
  float* numv    = (float*)carve((size_t)MROWS * 4);
  float* llv     = (float*)carve((size_t)MROWS * 4);

  // 1) pack/convert
  k_cvt_f16<<<2048, 256, 0, stream>>>(input, inph, MROWS * DD);
  k_pack_WT<<<dim3(640, 2), 256, 0, stream>>>(W_h, W_d, WhT, WdT);
  k_pack_WbiT<<<4096, 256, 0, stream>>>(W_bi, WpT);

  // 2) projections (WMMA): x = [relu(inp@W_h+b_h),1], y = [relu(inp@W_d+b_d),1]
  k_proj_wmma<<<dim3(MROWS / 16, KP / 16, 2), 32, 0, stream>>>(inph, WhT, WdT, b_h, b_d, xh, yh);

  // 3) fused biaffine (TDM LDS staging + WMMA x2 phases) -> s_rel (b,x,y,l)
  k_biaffine<<<dim3(NN / 16, LL, BB), 256, 0, stream>>>(xh, yh, WpT, emis);

  // 4) in-place log_softmax over L
  k_logsoftmax<<<(MROWS * NN) / 8, 256, 0, stream>>>(emis);

  // 5) CRF numerator, forward scan, final loss
  k_numerator<<<MROWS / 8, 256, 0, stream>>>(emis, wam, labels, transv, startv, endv, numv);
  k_crf_forward<<<MROWS, 128, 0, stream>>>(emis, wam, transv, startv, endv, numv, llv);
  k_finalize<<<1, 256, 0, stream>>>(llv, out);
}
